// ConductivityPredictor_48120813584643
// MI455X (gfx1250) — matrix operand order repaired and checked by reading, not compiled, not run on hardware
//
#include <hip/hip_runtime.h>

// ---------------------------------------------------------------------------
// GNN forward for MI455X (gfx1250, wave32, WMMA).
// Gather commuted through the dense layer => all GEMMs are 20000-row.
// bf16 data path (f32 accumulate in WMMA), CSR gather-reduce aggregation.
// Weight staging uses GLOBAL_LOAD_ASYNC_TO_LDS_B128 (ASYNCcnt DMA path).
// ---------------------------------------------------------------------------

#define N_NODES  20000
#define N_EDGES  320000
#define CHANNELS 256
#define IN_FEAT  118
#define KPAD     128   // IN_FEAT padded to a multiple of 32
#define LAYERS   4

typedef __attribute__((ext_vector_type(16))) __bf16 v16bf;
typedef __attribute__((ext_vector_type(8)))  float  v8f;
typedef __attribute__((address_space(3))) unsigned short lds_us;

__device__ __forceinline__ unsigned short f2bf_bits(float f) {
  union { float f; unsigned u; } a; a.f = f;
  unsigned r = a.u + 0x7FFFu + ((a.u >> 16) & 1u);   // round-to-nearest-even
  return (unsigned short)(r >> 16);
}
__device__ __forceinline__ float bf2f_bits(unsigned short h) {
  union { unsigned u; float f; } a; a.u = ((unsigned)h) << 16;
  return a.f;
}
__device__ __forceinline__ float gelu_exact(float x) {
  return 0.5f * x * (1.0f + erff(x * 0.70710678118654752f));
}

// ---------------------------------------------------------------------------
// CSR construction
// ---------------------------------------------------------------------------
__global__ void zero_u32(unsigned* p, int n) {
  int i = blockIdx.x * blockDim.x + threadIdx.x;
  if (i < n) p[i] = 0u;
}

__global__ void count_kernel(const long long* __restrict__ ei, unsigned* __restrict__ cnt) {
  int e = blockIdx.x * blockDim.x + threadIdx.x;
  if (e >= N_EDGES) return;
  atomicAdd(&cnt[(unsigned)ei[N_EDGES + e]], 1u);
}

__global__ void scan_kernel(const unsigned* __restrict__ cnt,
                            unsigned* __restrict__ rowptr,
                            unsigned* __restrict__ cursor,
                            float* __restrict__ inv_cnt) {
  __shared__ unsigned sh[1024];
  __shared__ unsigned carry;
  if (threadIdx.x == 0) carry = 0u;
  __syncthreads();
  for (int base = 0; base < N_NODES; base += 1024) {
    int i = base + (int)threadIdx.x;
    unsigned v = (i < N_NODES) ? cnt[i] : 0u;
    sh[threadIdx.x] = v;
    __syncthreads();
    for (int off = 1; off < 1024; off <<= 1) {
      unsigned t = (threadIdx.x >= (unsigned)off) ? sh[threadIdx.x - off] : 0u;
      __syncthreads();
      sh[threadIdx.x] += t;
      __syncthreads();
    }
    unsigned excl = carry + sh[threadIdx.x] - v;
    if (i < N_NODES) {
      rowptr[i] = excl;
      cursor[i] = excl;
      inv_cnt[i] = 1.0f / fmaxf((float)v, 1.0f);
    }
    __syncthreads();
    if (threadIdx.x == 0) carry += sh[1023];
    __syncthreads();
  }
  if (threadIdx.x == 0) rowptr[N_NODES] = carry;
}

__global__ void fill_kernel(const long long* __restrict__ ei,
                            unsigned* __restrict__ cursor,
                            unsigned* __restrict__ srcs) {
  int e = blockIdx.x * blockDim.x + threadIdx.x;
  if (e >= N_EDGES) return;
  unsigned s = (unsigned)ei[e];
  unsigned d = (unsigned)ei[N_EDGES + e];
  unsigned pos = atomicAdd(&cursor[d], 1u);
  srcs[pos] = s;
}

// ---------------------------------------------------------------------------
// Precision conversion / weight transpose (B operand must be column-major)
// ---------------------------------------------------------------------------
__global__ void cvt_x(const float* __restrict__ x, unsigned short* __restrict__ xb) {
  int idx = blockIdx.x * blockDim.x + threadIdx.x;
  if (idx >= N_NODES * KPAD) return;
  int n = idx / KPAD, k = idx - n * KPAD;
  xb[idx] = f2bf_bits(k < IN_FEAT ? x[(size_t)n * IN_FEAT + k] : 0.0f);
}

// out[n*Kdst + k] = bf16(W[k*256 + n]) (zero-pad k >= Ksrc)
__global__ void wtrans(const float* __restrict__ W, unsigned short* __restrict__ out,
                       int Ksrc, int Kdst) {
  int idx = blockIdx.x * blockDim.x + threadIdx.x;
  if (idx >= 256 * Kdst) return;
  int n = idx / Kdst, k = idx - n * Kdst;
  out[idx] = f2bf_bits(k < Ksrc ? W[(size_t)k * 256 + n] : 0.0f);
}

// ---------------------------------------------------------------------------
// GEMM: C[M x 256] = act(A[M x K](bf16) @ Wt^T + bias), output bf16.
// Per block: 8 waves x 16 rows = 128 rows; each wave owns a 16x256 strip
// (16 WMMA accumulator tiles). Wt staged through LDS in 64-wide K chunks
// via async global->LDS DMA; A and B fragments double-buffered so ds/vmem
// latency is covered by in-flight WMMAs.
// ---------------------------------------------------------------------------
#define KC 64
#define LDS_PITCH (KC + 8)   // pad 16 B per row -> conflict-free ds_load_b128

union BFrag { uint4 u[2]; v16bf v; };

__global__ void __launch_bounds__(256)
gemm_bias_act(const unsigned short* __restrict__ A,
              const unsigned short* __restrict__ Wt,
              const float* __restrict__ bias,
              unsigned short* __restrict__ Cout,
              int M, int K, int act) {
  __shared__ unsigned short lds[256 * LDS_PITCH];   // 36 KB

  const int wave = threadIdx.x >> 5;
  const int lane = threadIdx.x & 31;
  const int kh   = lane >> 4;   // which K-half of the fragment this lane holds
  const int lm   = lane & 15;

  const int rowBase = blockIdx.x * 128 + wave * 16;
  int arow = rowBase + lm;
  if (arow > M - 1) arow = M - 1;            // clamp; guarded at store
  const unsigned short* Arow = A + (size_t)arow * K;

  v8f acc[16];
  #pragma unroll
  for (int t = 0; t < 16; ++t) {
    #pragma unroll
    for (int j = 0; j < 8; ++j) acc[t][j] = 0.0f;
  }

  // Prefetch first A fragment (K-step 0).
  BFrag a_cur, a_nxt;
  a_cur.u[0] = *(const uint4*)(Arow + 8 * kh);
  a_cur.u[1] = *(const uint4*)(Arow + 16 + 8 * kh);

  const int total_ksteps = K >> 5;
  int step = 0;

  for (int kb = 0; kb < K; kb += KC) {
    __syncthreads();   // previous chunk's reads done before overwrite
    // Async DMA stage of Wt[:, kb..kb+64) -> LDS (256 rows x 64 bf16).
    for (int idx = threadIdx.x; idx < 256 * (KC / 8); idx += 256) {
      int n  = idx >> 3;              // row (output column index)
      int kc = (idx & 7) << 3;        // 8-element chunk within the K slice
      unsigned lofs = (unsigned)(unsigned long long)(lds_us*)(lds + n * LDS_PITCH + kc);
      const unsigned short* g = Wt + (size_t)n * K + kb + kc;
      asm volatile("global_load_async_to_lds_b128 %0, %1, off"
                   :: "v"(lofs), "v"(g) : "memory");
    }
    asm volatile("s_wait_asynccnt 0x0" ::: "memory");
    __syncthreads();

    #pragma unroll
    for (int ks = 0; ks < KC / 32; ++ks, ++step) {
      const int k0 = kb + ks * 32;
      // Prefetch next K-step's A fragment (global, L2-resident) early so the
      // loadcnt wait lands a full 16-WMMA loop later.
      if (step + 1 < total_ksteps) {
        const int k0n = k0 + 32;
        a_nxt.u[0] = *(const uint4*)(Arow + k0n + 8 * kh);
        a_nxt.u[1] = *(const uint4*)(Arow + k0n + 16 + 8 * kh);
      }

      // B fragments double-buffered: load nt+1 before issuing WMMA nt.
      BFrag b_cur, b_nxt;
      {
        const unsigned short* bp = lds + lm * LDS_PITCH + (k0 - kb) + 8 * kh;
        b_cur.u[0] = *(const uint4*)(bp);
        b_cur.u[1] = *(const uint4*)(bp + 16);
      }
      #pragma unroll
      for (int nt = 0; nt < 16; ++nt) {
        if (nt + 1 < 16) {
          const unsigned short* bp =
              lds + ((nt + 1) * 16 + lm) * LDS_PITCH + (k0 - kb) + 8 * kh;
          b_nxt.u[0] = *(const uint4*)(bp);
          b_nxt.u[1] = *(const uint4*)(bp + 16);
        }
        acc[nt] = __builtin_amdgcn_wmma_f32_16x16x32_bf16(
            false, a_cur.v, false, b_cur.v, (short)0, acc[nt], false, false);
        b_cur = b_nxt;
      }
      a_cur = a_nxt;
    }
  }

  // Epilogue: bias + exact GELU + bf16 store.
  // D layout: VGPR v -> row v + 8*kh (within tile), col = lm.
  #pragma unroll
  for (int nt = 0; nt < 16; ++nt) {
    const int col = nt * 16 + lm;
    const float bv = bias[col];
    #pragma unroll
    for (int v = 0; v < 8; ++v) {
      const int m = rowBase + kh * 8 + v;
      if (m < M) {
        float xv = acc[nt][v] + bv;
        if (act) xv = gelu_exact(xv);
        Cout[(size_t)m * CHANNELS + col] = f2bf_bits(xv);
      }
    }
  }
}

// ---------------------------------------------------------------------------
// Mean-aggregate over in-edges (CSR gather-reduce): one wave per node,
// each lane owns 8 channels (256/32). f32 accumulate, scale by 1/deg.
// ---------------------------------------------------------------------------
__global__ void __launch_bounds__(256)
agg_mean(const unsigned short* __restrict__ tb,
         const unsigned* __restrict__ rowptr,
         const unsigned* __restrict__ srcs,
         const float* __restrict__ inv_cnt,
         unsigned short* __restrict__ aggb) {
  int node = blockIdx.x * 8 + (threadIdx.x >> 5);
  if (node >= N_NODES) return;
  int lane = threadIdx.x & 31;
  unsigned beg = rowptr[node], end = rowptr[node + 1];
  float acc[8];
  #pragma unroll
  for (int j = 0; j < 8; ++j) acc[j] = 0.0f;
  for (unsigned i = beg; i < end; ++i) {
    unsigned s = srcs[i];
    uint4 v = *(const uint4*)(tb + (size_t)s * CHANNELS + lane * 8);
    const unsigned short* pv = (const unsigned short*)&v;
    #pragma unroll
    for (int j = 0; j < 8; ++j) acc[j] += bf2f_bits(pv[j]);
  }
  float w = inv_cnt[node];
  unsigned short ov[8];
  #pragma unroll
  for (int j = 0; j < 8; ++j) ov[j] = f2bf_bits(acc[j] * w);
  *(uint4*)(aggb + (size_t)node * CHANNELS + lane * 8) = *(const uint4*)ov;
}

// ---------------------------------------------------------------------------
// out[n] = mean_c h[n][c]; one wave per node, shuffle reduction.
// ---------------------------------------------------------------------------
__global__ void __launch_bounds__(256)
mean_kernel(const unsigned short* __restrict__ hb, float* __restrict__ out) {
  int node = blockIdx.x * 8 + (threadIdx.x >> 5);
  if (node >= N_NODES) return;
  int lane = threadIdx.x & 31;
  uint4 v = *(const uint4*)(hb + (size_t)node * CHANNELS + lane * 8);
  const unsigned short* pv = (const unsigned short*)&v;
  float s = 0.0f;
  #pragma unroll
  for (int j = 0; j < 8; ++j) s += bf2f_bits(pv[j]);
  for (int off = 16; off > 0; off >>= 1) s += __shfl_down(s, off, 32);
  if (lane == 0) out[node] = s * (1.0f / 256.0f);
}

// ---------------------------------------------------------------------------
extern "C" void kernel_launch(void* const* d_in, const int* in_sizes, int n_in,
                              void* d_out, int out_size, void* d_ws, size_t ws_size,
                              hipStream_t stream) {
  (void)in_sizes; (void)n_in; (void)out_size; (void)ws_size;

  const float*     x       = (const float*)d_in[0];
  const long long* ei      = (const long long*)d_in[1];
  const float*     W_embed = (const float*)d_in[2];
  const float*     b_embed = (const float*)d_in[3];
  const float*     W1      = (const float*)d_in[4];
  const float*     b1      = (const float*)d_in[5];
  const float*     W2      = (const float*)d_in[6];
  const float*     b2      = (const float*)d_in[7];
  float*           out     = (float*)d_out;

  // Workspace carve-out (all 256 B aligned), ~38.5 MB total.
  char* p = (char*)d_ws;
  auto take = [&](size_t bytes) -> char* {
    char* r = p; p += (bytes + 255) & ~(size_t)255; return r;
  };
  unsigned*       cnt     = (unsigned*)take((size_t)N_NODES * 4);
  float*          inv_cnt = (float*)take((size_t)N_NODES * 4);
  unsigned*       rowptr  = (unsigned*)take((size_t)(N_NODES + 1) * 4);
  unsigned*       cursor  = (unsigned*)take((size_t)N_NODES * 4);
  unsigned*       srcs    = (unsigned*)take((size_t)N_EDGES * 4);
  unsigned short* xb      = (unsigned short*)take((size_t)N_NODES * KPAD * 2);
  unsigned short* WtE     = (unsigned short*)take((size_t)256 * KPAD * 2);
  unsigned short* Wt1     = (unsigned short*)take((size_t)LAYERS * 256 * 256 * 2);
  unsigned short* Wt2     = (unsigned short*)take((size_t)LAYERS * 256 * 256 * 2);
  unsigned short* hb      = (unsigned short*)take((size_t)N_NODES * 256 * 2);
  unsigned short* tb      = (unsigned short*)take((size_t)N_NODES * 256 * 2);
  unsigned short* aggb    = (unsigned short*)take((size_t)N_NODES * 256 * 2);

  // 1) CSR by destination + inverse degree (reused across all 4 layers)
  zero_u32<<<(N_NODES + 255) / 256, 256, 0, stream>>>(cnt, N_NODES);
  count_kernel<<<(N_EDGES + 255) / 256, 256, 0, stream>>>(ei, cnt);
  scan_kernel<<<1, 1024, 0, stream>>>(cnt, rowptr, cursor, inv_cnt);
  fill_kernel<<<(N_EDGES + 255) / 256, 256, 0, stream>>>(ei, cursor, srcs);

  // 2) bf16 conversions; weights transposed for column-major B fragments
  cvt_x<<<(N_NODES * KPAD + 255) / 256, 256, 0, stream>>>(x, xb);
  wtrans<<<(256 * KPAD + 255) / 256, 256, 0, stream>>>(W_embed, WtE, IN_FEAT, KPAD);
  for (int l = 0; l < LAYERS; ++l) {
    wtrans<<<(256 * 256 + 255) / 256, 256, 0, stream>>>(
        W1 + (size_t)l * 256 * 256, Wt1 + (size_t)l * 256 * 256, 256, 256);
    wtrans<<<(256 * 256 + 255) / 256, 256, 0, stream>>>(
        W2 + (size_t)l * 256 * 256, Wt2 + (size_t)l * 256 * 256, 256, 256);
  }

  const int gblocks = (N_NODES + 127) / 128;

  // 3) embed: h = x @ W_embed + b (no activation)
  gemm_bias_act<<<gblocks, 256, 0, stream>>>(xb, WtE, b_embed, hb, N_NODES, KPAD, 0);

  // 4) layers: t = gelu(h@W1+b1); agg = CSR-mean(t[src]); h = gelu(agg@W2+b2)
  for (int l = 0; l < LAYERS; ++l) {
    gemm_bias_act<<<gblocks, 256, 0, stream>>>(
        hb, Wt1 + (size_t)l * 65536, b1 + l * 256, tb, N_NODES, 256, 1);
    agg_mean<<<(N_NODES + 7) / 8, 256, 0, stream>>>(tb, rowptr, srcs, inv_cnt, aggb);
    gemm_bias_act<<<gblocks, 256, 0, stream>>>(
        aggb, Wt2 + (size_t)l * 65536, b2 + l * 256, hb, N_NODES, 256, 1);
  }

  // 5) out = mean over channels
  mean_kernel<<<(N_NODES + 7) / 8, 256, 0, stream>>>(hb, out);
}
